// ValueNetwork_74689481277646
// MI455X (gfx1250) — matrix operand order repaired and checked by reading, not compiled, not run on hardware
//
#include <hip/hip_runtime.h>

typedef __attribute__((ext_vector_type(16))) _Float16 v16h;
typedef __attribute__((ext_vector_type(8)))  _Float16 v8h;
typedef __attribute__((ext_vector_type(8)))  float    v8f;
typedef __attribute__((ext_vector_type(2)))  float    v2f;

// Shapes: B=256, N=511, IN=13, SELF=6, HS=7(pad 8), rows=512, D1=64
__global__ __launch_bounds__(256)
void value_net_kernel(const float* __restrict__ state,
                      const float* __restrict__ tmat,
                      const float* __restrict__ w_a,
                      const float* __restrict__ w1,
                      const float* __restrict__ w2,
                      const float* __restrict__ l1_w,
                      const float* __restrict__ l1_b,
                      const float* __restrict__ l2_w,
                      const float* __restrict__ l2_b,
                      float* __restrict__ out)
{
    __shared__ __align__(16) float    Xs[512 * 8];      // X, padded col7 = 0
    __shared__ __align__(16) _Float16 XW1t[64 * 512];   // (X@w1)^T [d][m], f16
    __shared__ __align__(16) _Float16 Et[32 * 512];     // exp(l - rowmax) tile, f16
    __shared__ __align__(16) float    Xwt[32 * 8];      // (X@w_a) rows of current tile
    __shared__ float rmax2[32 * 64];                    // per-(row, colgroup) maxima
    __shared__ float rsum2[32 * 64];                    // per-(row, colgroup) exp sums
    __shared__ float rowM[32];
    __shared__ float SinvAll[512];                      // 1/rowsum for every row
    __shared__ float p0u[512];                          // unnormalized e of row 0
    __shared__ float vacc[64];                          // sum_m e0[m]*Sinv[m]*relu(h1[m,:])
    __shared__ float wa_s[49];
    __shared__ float w1_s[7 * 64];
    __shared__ float f2[64];
    __shared__ float hid[64];

    const int tid = threadIdx.x;
    const int b   = blockIdx.x;

    if (tid < 64) vacc[tid] = 0.0f;
    if (tid < 49) wa_s[tid] = w_a[tid];
    for (int i = tid; i < 448; i += 256) w1_s[i] = w1[i];

    // ---------- Build X (row 0 = self_state @ t, rows 1..511 = human states)
    for (int m = tid; m < 512; m += 256) {
        if (m == 0) {
            const float* ss = state + (size_t)b * 511 * 13;
            for (int j = 0; j < 7; ++j) {
                float s = 0.f;
                for (int i = 0; i < 6; ++i) s += ss[i] * tmat[i * 7 + j];
                Xs[j] = s;
            }
            Xs[7] = 0.f;
        } else {
            const float* src = state + ((size_t)b * 511 + (m - 1)) * 13 + 6;
            for (int j = 0; j < 7; ++j) Xs[m * 8 + j] = src[j];
            Xs[m * 8 + 7] = 0.f;
        }
    }
    __syncthreads();

    // ---------- XW1t = (X @ w1)^T in f16
    for (int idx = tid; idx < 512 * 64; idx += 256) {
        const int m = idx & 511, d = idx >> 9;
        float s = 0.f;
        for (int j = 0; j < 7; ++j) s += Xs[m * 8 + j] * w1_s[j * 64 + d];
        XW1t[d * 512 + m] = (_Float16)s;
    }
    __syncthreads();

    const int lane = tid & 31;
    const int wave = tid >> 5;
    const int hi   = lane >> 4;      // lane half (selects K sub-range / row half)
    const int lan  = lane & 15;
    const int rsub = wave >> 2;      // 16-row sub-tile within the 32-row tile
    const int grp  = wave & 3;       // column group (logits) / output col tile (h1)

    for (int nt = 0; nt < 16; ++nt) {
        // ---- Xw rows for this 32-row tile (K padded to 8 with zero)
        {
            const int row = tid >> 3;
            const int k   = tid & 7;
            const int n   = nt * 32 + row;
            float s = 0.f;
            if (k < 7) {
                for (int j = 0; j < 7; ++j) s += Xs[n * 8 + j] * wa_s[j * 7 + k];
            }
            Xwt[row * 8 + k] = s;
        }
        __syncthreads();

        // ---- logits via f32 WMMA: D tiles (16x16) = Xw(16x8) @ X^T(8x16), kept in regs
        // A lane layout (16x4 f32): M = lan, K = hi*2 + {0,1};  B = X^T mirrors it.
        const v2f aLo = *(const v2f*)&Xwt[(rsub * 16 + lan) * 8 + hi * 2];
        const v2f aHi = *(const v2f*)&Xwt[(rsub * 16 + lan) * 8 + hi * 2 + 4];
        v8f D[8];
        #pragma unroll
        for (int i = 0; i < 8; ++i) {
            const int m0 = (grp + 4 * i) * 16;
            const v2f bLo = *(const v2f*)&Xs[(m0 + lan) * 8 + hi * 2];
            const v2f bHi = *(const v2f*)&Xs[(m0 + lan) * 8 + hi * 2 + 4];
            v8f z = {};
            v8f d0 = __builtin_amdgcn_wmma_f32_16x16x4_f32(
                         false, aLo, false, bLo, (short)0, z, false, false);
            D[i]   = __builtin_amdgcn_wmma_f32_16x16x4_f32(
                         false, aHi, false, bHi, (short)0, d0, false, false);
        }

        // ---- row max: per-lane partial over 8 tiles, then 64-wide reduce
        #pragma unroll
        for (int j = 0; j < 8; ++j) {
            const int rrel = rsub * 16 + hi * 8 + j;     // D layout: row = j + hi*8
            float mx = D[0][j];
            #pragma unroll
            for (int i = 1; i < 8; ++i) mx = fmaxf(mx, D[i][j]);
            rmax2[rrel * 64 + grp * 16 + lan] = mx;
        }
        __syncthreads();
        if (tid < 32) {
            float M = rmax2[tid * 64];
            for (int i = 1; i < 64; ++i) M = fmaxf(M, rmax2[tid * 64 + i]);
            rowM[tid] = M;
        }
        __syncthreads();

        // ---- e = exp(l - M): write f16 E tile, partial sums, capture row-0 e's
        #pragma unroll
        for (int j = 0; j < 8; ++j) {
            const int rrel = rsub * 16 + hi * 8 + j;
            const float M = rowM[rrel];
            float s = 0.f;
            #pragma unroll
            for (int i = 0; i < 8; ++i) {
                const int m = (grp + 4 * i) * 16 + lan;
                const float e = __expf(D[i][j] - M);
                Et[rrel * 512 + m] = (_Float16)e;
                if (nt == 0 && rrel == 0) p0u[m] = e;
                s += e;
            }
            rsum2[rrel * 64 + grp * 16 + lan] = s;
        }
        __syncthreads();
        if (tid < 32) {
            float S = 0.f;
            for (int i = 0; i < 64; ++i) S += rsum2[tid * 64 + i];
            SinvAll[nt * 32 + tid] = 1.0f / S;
        }
        __syncthreads();

        // ---- f16 WMMA: 16x16 tile of E(32x512) @ XW1(512x64), K chained over 16
        v8f acc = {};
        const _Float16* pA = &Et[(rsub * 16 + lan) * 512 + hi * 8];
        const _Float16* pB = &XW1t[(grp * 16 + lan) * 512 + hi * 8];
        #pragma unroll
        for (int kt = 0; kt < 16; ++kt) {
            const int k0 = kt * 32;
            v8h alo = *(const v8h*)(pA + k0);
            v8h ahi = *(const v8h*)(pA + k0 + 16);
            v8h blo = *(const v8h*)(pB + k0);
            v8h bhi = *(const v8h*)(pB + k0 + 16);
            v16h Am = __builtin_shufflevector(alo, ahi, 0,1,2,3,4,5,6,7,8,9,10,11,12,13,14,15);
            v16h Bm = __builtin_shufflevector(blo, bhi, 0,1,2,3,4,5,6,7,8,9,10,11,12,13,14,15);
            acc = __builtin_amdgcn_wmma_f32_16x16x32_f16(
                      false, Am, false, Bm, (short)0, acc, false, false);
        }
        // relu + deferred-softmax weighting: h1[m,:] = Sinv[m]*relu(raw),
        // weight by p0u[m]; global Sinv[0] applied in the epilogue.
        {
            const int dcol = grp * 16 + lan;
            const int mrow = nt * 32 + rsub * 16 + hi * 8;
            float contrib = 0.f;
            #pragma unroll
            for (int j = 0; j < 8; ++j) {
                const float h = fmaxf(acc[j], 0.f);
                contrib += p0u[mrow + j] * SinvAll[mrow + j] * h;
            }
            atomicAdd(&vacc[dcol], contrib);
        }
        __syncthreads();
    }

    // ---------- epilogue
    if (tid < 64) {
        const float s0 = SinvAll[0];
        float s = 0.f;
        for (int k = 0; k < 64; ++k) s += vacc[k] * w2[k * 64 + tid];
        f2[tid] = fmaxf(s * s0, 0.f);
    }
    __syncthreads();
    if (tid < 64) {
        float s = l1_b[tid];
        for (int k = 0; k < 64; ++k) s += f2[k] * l1_w[tid * 64 + k];
        hid[tid] = fmaxf(s, 0.f);
    }
    __syncthreads();
    if (tid == 0) {
        float s = l2_b[0];
        for (int k = 0; k < 64; ++k) s += hid[k] * l2_w[k];
        out[b] = s;
    }
}

extern "C" void kernel_launch(void* const* d_in, const int* in_sizes, int n_in,
                              void* d_out, int out_size, void* d_ws, size_t ws_size,
                              hipStream_t stream) {
    const float* state = (const float*)d_in[0];
    const float* tmat  = (const float*)d_in[1];
    const float* w_a   = (const float*)d_in[2];
    const float* w1    = (const float*)d_in[3];
    const float* w2    = (const float*)d_in[4];
    const float* l1_w  = (const float*)d_in[5];
    const float* l1_b  = (const float*)d_in[6];
    const float* l2_w  = (const float*)d_in[7];
    const float* l2_b  = (const float*)d_in[8];
    float* out = (float*)d_out;

    value_net_kernel<<<dim3(256), dim3(256), 0, stream>>>(
        state, tmat, w_a, w1, w2, l1_w, l1_b, l2_w, l2_b, out);
}
